// GromovWassersteinLoss_25615184953657
// MI455X (gfx1250) — matrix operand order repaired and checked by reading, not compiled, not run on hardware
//
#include <hip/hip_runtime.h>

// ---------------------------------------------------------------------------
// Entropic Gromov-Wasserstein on MI455X (gfx1250).
// Core GEMMs: v_wmma_f32_16x16x32_bf16 (bf16 in, f32 accumulate), with
// double-buffered global_load_async_to_lds_b128 tile staging (ASYNCcnt).
// Sinkhorn runs as a persistent kernel with a software grid barrier.
// ---------------------------------------------------------------------------

#define N_PTS   2048
#define D_EMB   768
#define D_HYP   65
#define SINK_ITERS 50
#define GW_ITERS   20
#define SINK_BLOCKS 64

typedef __attribute__((ext_vector_type(16))) __bf16 bf16x16;
typedef __attribute__((ext_vector_type(8)))  float  f32x8;

__device__ __forceinline__ unsigned short f2bf(float f) {
  unsigned int u = __float_as_uint(f);
  unsigned int r = u + 0x7FFFu + ((u >> 16) & 1u);   // round-to-nearest-even
  return (unsigned short)(r >> 16);
}
__device__ __forceinline__ float bf2f(unsigned short h) {
  return __uint_as_float(((unsigned int)h) << 16);
}
__device__ __forceinline__ float wave_sum(float x) {
#pragma unroll
  for (int o = 16; o > 0; o >>= 1) x += __shfl_xor(x, o, 32);
  return x;
}
__device__ __forceinline__ float wave_max(float x) {
#pragma unroll
  for (int o = 16; o > 0; o >>= 1) x = fmaxf(x, __shfl_xor(x, o, 32));
  return x;
}

// CDNA5 async global->LDS copy: each lane moves 32 bytes (2 x b128).
// INST_OFFSET is added to both LDS and global addresses (ISA 15.18.3 op 98).
__device__ __forceinline__ void async_b128x2(unsigned int lds_off,
                                             const unsigned short* g) {
  unsigned long long ga = (unsigned long long)g;
  asm volatile("global_load_async_to_lds_b128 %0, %1, off\n\t"
               "global_load_async_to_lds_b128 %0, %1, off offset:16"
               :: "v"(lds_off), "v"(ga) : "memory");
}
__device__ __forceinline__ void wait_async0() {
  asm volatile("s_wait_asynccnt 0x0" ::: "memory");
}

// ---------------------------------------------------------------------------
// BF16 WMMA GEMM: D = A(MxK row-major) @ Bt^T, with Bt given N x K row-major
// (i.e. Bt[n][k] = B[k][n]).  M = N = 2048.  Block tile 128x128, K-step 32,
// 8 wave32 waves (each 32x64 of output).  Fragments per ISA 7.12.2 tables.
// Tiles staged with double-buffered async-to-LDS copies.
// Epilogue modes:
//   0 GRAM  : d = (i==j)?0:sqrt(max(sq_i+sq_j-2g,1e-12)) if >0; max-reduce
//   1 STORET: store bf16(g) TRANSPOSED (out[col][row]) for the next GEMM's Bt
//   2 SINK  : tens=2(a_i+b_j)-4g; K=exp(-tens/eps); store K and K^T (bf16)
//   3 OBJ   : tens=(a_i+b_j)-2g;  accumulate tens*T -> per-block partial
// ---------------------------------------------------------------------------
#define MTL 128
#define NTL 128
#define KTS 32
#define LPAD 40   // ushorts per LDS row (80B rows keep 16B-aligned chunks)

constexpr int MODE_GRAM = 0, MODE_STORET = 1, MODE_SINK = 2, MODE_OBJ = 3;

template<int MODE>
__global__ __launch_bounds__(256)
void gemm_bf16(const unsigned short* __restrict__ A,
               const unsigned short* __restrict__ Bt,
               int K,
               float* __restrict__ outF,
               unsigned short* __restrict__ outB,
               unsigned short* __restrict__ outBT,
               const float* __restrict__ av,
               const float* __restrict__ bv,
               const float* __restrict__ sq,
               const float* __restrict__ Tf,
               float* __restrict__ objPart,
               unsigned int* __restrict__ maxBits)
{
  __shared__ unsigned short lA[2][MTL * LPAD];   // [m][k]
  __shared__ unsigned short lB[2][NTL * LPAD];   // [n][k]
  __shared__ float wred[8];

  const int tid  = threadIdx.x;
  const int lane = tid & 31;
  const int wave = tid >> 5;
  const int wm = wave & 3;       // 4 waves along M (32 rows each)
  const int wn = wave >> 2;      // 2 waves along N (64 cols each)
  const int bM = blockIdx.y * MTL;
  const int bN = blockIdx.x * NTL;

  f32x8 acc[2][4];
#pragma unroll
  for (int mt = 0; mt < 2; ++mt)
#pragma unroll
    for (int nt = 0; nt < 4; ++nt)
#pragma unroll
      for (int e = 0; e < 8; ++e) acc[mt][nt][e] = 0.0f;

  // staging: each thread moves one 32B row-chunk of A and one of Bt per tile
  const int arow  = tid >> 1;            // 0..127
  const int akoff = (tid & 1) * 16;      // 0 or 16 (ushorts)
  const unsigned short* aG = A  + (size_t)(bM + arow) * K + akoff;
  const unsigned short* bG = Bt + (size_t)(bN + arow) * K + akoff;
  const unsigned int laOff0 = (unsigned int)(unsigned long long)&lA[0][arow * LPAD + akoff];
  const unsigned int laOff1 = (unsigned int)(unsigned long long)&lA[1][arow * LPAD + akoff];
  const unsigned int lbOff0 = (unsigned int)(unsigned long long)&lB[0][arow * LPAD + akoff];
  const unsigned int lbOff1 = (unsigned int)(unsigned long long)&lB[1][arow * LPAD + akoff];

  const int nK = K / KTS;
  // stage tile 0 into buffer 0
  async_b128x2(laOff0, aG);
  async_b128x2(lbOff0, bG);
  wait_async0();
  __syncthreads();

  const int khalf = lane >> 4;
  const int lmod  = lane & 15;

  for (int kt = 0; kt < nK; ++kt) {
    const int cur = kt & 1;
    if (kt + 1 < nK) {   // async-prefetch next K-slab into the other buffer
      const unsigned short* an = aG + (size_t)(kt + 1) * KTS;
      const unsigned short* bn = bG + (size_t)(kt + 1) * KTS;
      async_b128x2(cur ? laOff0 : laOff1, an);
      async_b128x2(cur ? lbOff0 : lbOff1, bn);
    }

    // fragment loads (two 16B LDS reads each) + 8 WMMAs per wave
    bf16x16 af[2], bfm[4];
#pragma unroll
    for (int mt = 0; mt < 2; ++mt) {
      // A lane: row=lmod; K = {khalf*8..+7} then {16+khalf*8..+7}
      const unsigned short* s = &lA[cur][(wm * 32 + mt * 16 + lmod) * LPAD + khalf * 8];
      uint4* d = reinterpret_cast<uint4*>(&af[mt]);
      d[0] = *reinterpret_cast<const uint4*>(s);
      d[1] = *reinterpret_cast<const uint4*>(s + 16);
    }
#pragma unroll
    for (int nt = 0; nt < 4; ++nt) {
      // B lane: col=lmod; K = {khalf*8..+7} then {16+khalf*8..+7}
      const unsigned short* s = &lB[cur][(wn * 64 + nt * 16 + lmod) * LPAD + khalf * 8];
      uint4* d = reinterpret_cast<uint4*>(&bfm[nt]);
      d[0] = *reinterpret_cast<const uint4*>(s);
      d[1] = *reinterpret_cast<const uint4*>(s + 16);
    }
#pragma unroll
    for (int mt = 0; mt < 2; ++mt)
#pragma unroll
      for (int nt = 0; nt < 4; ++nt)
        acc[mt][nt] = __builtin_amdgcn_wmma_f32_16x16x32_bf16(
            false, af[mt], false, bfm[nt], (short)0, acc[mt][nt], false, false);

    wait_async0();      // next tile fully landed in LDS (per-wave ASYNCcnt)
    __syncthreads();    // everyone done reading cur + everyone's async done
  }

  // ---- epilogue ----
  float lred = 0.0f;   // max (GRAM) or sum (OBJ)
#pragma unroll
  for (int mt = 0; mt < 2; ++mt) {
#pragma unroll
    for (int nt = 0; nt < 4; ++nt) {
      float* ap = reinterpret_cast<float*>(&acc[mt][nt]);
      const int col = bN + wn * 64 + nt * 16 + lmod;
#pragma unroll
      for (int r = 0; r < 8; ++r) {
        const int row = bM + wm * 32 + mt * 16 + khalf * 8 + r;
        const float g = ap[r];
        const size_t idx = (size_t)row * N_PTS + col;
        if (MODE == MODE_GRAM) {
          float d2 = sq[row] + sq[col] - 2.0f * g;
          float d  = (row == col) ? 0.0f
                     : ((d2 > 0.0f) ? sqrtf(fmaxf(d2, 1e-12f)) : 0.0f);
          outF[idx] = d;
          lred = fmaxf(lred, d);
        } else if (MODE == MODE_STORET) {
          outB[(size_t)col * N_PTS + row] = f2bf(g);   // transposed -> next Bt
        } else if (MODE == MODE_SINK) {
          float tens = 2.0f * (av[row] + bv[col]) - 4.0f * g;
          float kk = __expf(-tens * 10.0f);            // eps = 0.1
          unsigned short kb = f2bf(kk);
          outB[idx] = kb;
          outBT[(size_t)col * N_PTS + row] = kb;       // K^T for fast col matvec
        } else {                                       // MODE_OBJ
          float tens = (av[row] + bv[col]) - 2.0f * g;
          lred += tens * Tf[idx];
        }
      }
    }
  }
  if (MODE == MODE_GRAM) {
    float m = wave_max(lred);
    if (lane == 0) atomicMax(maxBits, __float_as_uint(m));   // exact, order-free
  } else if (MODE == MODE_OBJ) {
    float s = wave_sum(lred);
    if (lane == 0) wred[wave] = s;
    __syncthreads();
    if (tid == 0) {
      float t = 0.0f;
      for (int w = 0; w < 8; ++w) t += wred[w];
      objPart[blockIdx.y * gridDim.x + blockIdx.x] = t;      // deterministic
    }
  }
}

// ---------------------------------------------------------------------------
// Small kernels
// ---------------------------------------------------------------------------
__global__ void k_init(unsigned int* scal) {
  if (threadIdx.x == 0 && blockIdx.x == 0) {
    scal[0] = 0u;  // max (src cost)
    scal[1] = 0u;  // max (hyp cost)
    scal[2] = 0u;  // grid-barrier arrive count
    scal[3] = 0u;  // grid-barrier generation
  }
}

__global__ void k_cvt_src(const float* __restrict__ X,
                          unsigned short* __restrict__ Xb) {
  int idx = blockIdx.x * blockDim.x + threadIdx.x;
  if (idx < N_PTS * D_EMB) Xb[idx] = f2bf(X[idx]);
}

__global__ __launch_bounds__(256)
void k_rownorm(const float* __restrict__ X, float* __restrict__ sq) {
  __shared__ float red[8];
  const int row = blockIdx.x;
  float s = 0.0f;
  for (int c = threadIdx.x; c < D_EMB; c += 256) {
    float x = X[(size_t)row * D_EMB + c];
    s += x * x;
  }
  s = wave_sum(s);
  const int lane = threadIdx.x & 31, wave = threadIdx.x >> 5;
  if (lane == 0) red[wave] = s;
  __syncthreads();
  if (threadIdx.x == 0) {
    float t = 0.0f;
    for (int w = 0; w < 8; ++w) t += red[w];
    sq[row] = t;
  }
}

__global__ __launch_bounds__(256)
void k_hyp(const float* __restrict__ H, float* __restrict__ D,
           unsigned int* __restrict__ maxBits) {
  __shared__ float sI[16 * D_HYP];
  __shared__ float sJ[16 * D_HYP];
  __shared__ float red[8];
  const int i0 = blockIdx.y * 16, j0 = blockIdx.x * 16;
  for (int t = threadIdx.x; t < 16 * D_HYP; t += 256) {
    int r = t / D_HYP, c = t - r * D_HYP;
    sI[t] = H[(size_t)(i0 + r) * D_HYP + c];
    sJ[t] = H[(size_t)(j0 + r) * D_HYP + c];
  }
  __syncthreads();
  const int tx = threadIdx.x & 15, ty = threadIdx.x >> 4;
  float inner = -sI[ty * D_HYP] * sJ[tx * D_HYP];      // Lorentz signature
#pragma unroll 8
  for (int d = 1; d < D_HYP; ++d)
    inner += sI[ty * D_HYP + d] * sJ[tx * D_HYP + d];
  float z = fmaxf(-inner, 1.0f + 1e-7f);
  float dd = logf(z + sqrtf(z * z - 1.0f));            // arccosh(z), finite
  D[(size_t)(i0 + ty) * N_PTS + (j0 + tx)] = dd;
  float m = wave_max(dd);
  const int lane = threadIdx.x & 31, wave = threadIdx.x >> 5;
  if (lane == 0) red[wave] = m;
  __syncthreads();
  if (threadIdx.x == 0) {
    float t = red[0];
    for (int w = 1; w < 8; ++w) t = fmaxf(t, red[w]);
    atomicMax(maxBits, __float_as_uint(t));
  }
}

__global__ void k_normalize(const float* __restrict__ D,
                            const unsigned int* __restrict__ maxBits,
                            unsigned short* __restrict__ Cb) {
  const float mx = __uint_as_float(*maxBits);
  const float s = (mx > 1e-8f) ? 1.0f / (mx + 1e-8f) : 0.0f;
  int idx = blockIdx.x * blockDim.x + threadIdx.x;
  if (idx < N_PTS * N_PTS) Cb[idx] = f2bf(D[idx] * s);
}

__global__ __launch_bounds__(256)
void k_rowmeansq(const unsigned short* __restrict__ Cb, float* __restrict__ outv) {
  __shared__ float red[8];
  const int row = blockIdx.x;
  float s = 0.0f;
  for (int c = threadIdx.x; c < N_PTS; c += 256) {
    float x = bf2f(Cb[(size_t)row * N_PTS + c]);
    s += x * x;
  }
  s = wave_sum(s);
  const int lane = threadIdx.x & 31, wave = threadIdx.x >> 5;
  if (lane == 0) red[wave] = s;
  __syncthreads();
  if (threadIdx.x == 0) {
    float t = 0.0f;
    for (int w = 0; w < 8; ++w) t += red[w];
    outv[row] = t * (1.0f / (float)N_PTS);
  }
}

__global__ void k_fillT(unsigned short* __restrict__ Tb) {
  int idx = blockIdx.x * blockDim.x + threadIdx.x;
  if (idx < N_PTS * N_PTS) Tb[idx] = 0x3480;  // bf16(2^-22) == 1/(2048*2048) exactly
}

// software grid barrier (count + generation); all SINK_BLOCKS blocks resident
__device__ __forceinline__ void grid_barrier(volatile unsigned int* cnt,
                                             volatile unsigned int* gen,
                                             unsigned int nb) {
  __syncthreads();
  if (threadIdx.x == 0) {
    unsigned int g = *gen;
    __threadfence();
    if (atomicAdd((unsigned int*)cnt, 1u) == nb - 1u) {
      *cnt = 0u;
      __threadfence();
      atomicAdd((unsigned int*)gen, 1u);
    } else {
      while (*gen == g) { __builtin_amdgcn_s_sleep(2); }
    }
    __threadfence();
  }
  __syncthreads();
}

__device__ __forceinline__ float dot_row_bf16(const unsigned short* __restrict__ Kr,
                                              const float* __restrict__ vr) {
  float s = 0.0f;
#pragma unroll
  for (int ch = 0; ch < 8; ++ch) {
    uint4 pk = *reinterpret_cast<const uint4*>(Kr + ch * 8);
    unsigned int w0 = pk.x, w1 = pk.y, w2 = pk.z, w3 = pk.w;
    const float* vv = vr + ch * 8;
    s += bf2f((unsigned short)(w0 & 0xffffu)) * vv[0];
    s += bf2f((unsigned short)(w0 >> 16))     * vv[1];
    s += bf2f((unsigned short)(w1 & 0xffffu)) * vv[2];
    s += bf2f((unsigned short)(w1 >> 16))     * vv[3];
    s += bf2f((unsigned short)(w2 & 0xffffu)) * vv[4];
    s += bf2f((unsigned short)(w2 >> 16))     * vv[5];
    s += bf2f((unsigned short)(w3 & 0xffffu)) * vv[6];
    s += bf2f((unsigned short)(w3 >> 16))     * vv[7];
  }
  return s;
}

// persistent Sinkhorn: 50 iterations, one wave per row, grid barrier between halves
__global__ __launch_bounds__(256)
void k_sinkhorn(const unsigned short* __restrict__ Kb,
                const unsigned short* __restrict__ KTb,
                float* __restrict__ u, float* __restrict__ v,
                unsigned int* __restrict__ scal) {
  volatile unsigned int* cnt = scal + 2;
  volatile unsigned int* gen = scal + 3;
  const int lane = threadIdx.x & 31, wave = threadIdx.x >> 5;

  int g = blockIdx.x * 256 + threadIdx.x;
  if (g < N_PTS) { u[g] = 1.0f; v[g] = 1.0f; }
  grid_barrier(cnt, gen, SINK_BLOCKS);

  const float pq = 1.0f / (float)N_PTS;
  for (int it = 0; it < SINK_ITERS; ++it) {
    // u = p / (K @ v + 1e-16)
#pragma unroll
    for (int rr = 0; rr < 4; ++rr) {
      const int row = blockIdx.x * 32 + wave * 4 + rr;
      float s = dot_row_bf16(Kb + (size_t)row * N_PTS + lane * 64, v + lane * 64);
      s = wave_sum(s);
      if (lane == 0) u[row] = pq / (s + 1e-16f);
    }
    grid_barrier(cnt, gen, SINK_BLOCKS);
    // v = q / (K^T @ u + 1e-16)
#pragma unroll
    for (int rr = 0; rr < 4; ++rr) {
      const int row = blockIdx.x * 32 + wave * 4 + rr;
      float s = dot_row_bf16(KTb + (size_t)row * N_PTS + lane * 64, u + lane * 64);
      s = wave_sum(s);
      if (lane == 0) v[row] = pq / (s + 1e-16f);
    }
    grid_barrier(cnt, gen, SINK_BLOCKS);
  }
}

__global__ void k_formT(const unsigned short* __restrict__ Kb,
                        const float* __restrict__ u, const float* __restrict__ v,
                        unsigned short* __restrict__ Tb, float* __restrict__ Tf) {
  int idx = blockIdx.x * blockDim.x + threadIdx.x;
  if (idx < N_PTS * N_PTS) {
    int i = idx >> 11, j = idx & (N_PTS - 1);
    float t = u[i] * bf2f(Kb[idx]) * v[j];
    Tb[idx] = f2bf(t);
    Tf[idx] = t;
  }
}

__global__ void k_finalsum(const float* __restrict__ part, float* __restrict__ out) {
  if (blockIdx.x == 0 && threadIdx.x == 0) {
    float s = 0.0f;
    for (int i = 0; i < 256; ++i) s += part[i];   // fixed order -> deterministic
    out[0] = s;
  }
}

// ---------------------------------------------------------------------------
// Host orchestration
// ---------------------------------------------------------------------------
extern "C" void kernel_launch(void* const* d_in, const int* in_sizes, int n_in,
                              void* d_out, int out_size, void* d_ws, size_t ws_size,
                              hipStream_t stream) {
  (void)in_sizes; (void)n_in; (void)out_size; (void)ws_size;
  const float* X = (const float*)d_in[0];       // (2048, 768)  f32
  const float* H = (const float*)d_in[1];       // (2048, 65)   f32
  float* out = (float*)d_out;                   // scalar f32

  char* w = (char*)d_ws;
  size_t off = 0;
  auto alloc = [&](size_t bytes) -> void* {
    void* p = (void*)(w + off);
    off += (bytes + 255) & ~(size_t)255;
    return p;
  };
  const size_t NN = (size_t)N_PTS * N_PTS;
  unsigned short* Xb    = (unsigned short*)alloc((size_t)N_PTS * D_EMB * 2);
  unsigned short* C1b   = (unsigned short*)alloc(NN * 2);
  unsigned short* C2b   = (unsigned short*)alloc(NN * 2);
  unsigned short* Tb    = (unsigned short*)alloc(NN * 2);
  unsigned short* TC2t  = (unsigned short*)alloc(NN * 2);  // (T@C2^T)^T, N x K
  unsigned short* Kb    = (unsigned short*)alloc(NN * 2);
  unsigned short* KTb   = (unsigned short*)alloc(NN * 2);
  float* Df   = (float*)alloc(NN * 4);          // scratch f32: costs, then T_f32
  float* sq   = (float*)alloc(N_PTS * 4);
  float* av   = (float*)alloc(N_PTS * 4);
  float* bvv  = (float*)alloc(N_PTS * 4);
  float* u    = (float*)alloc(N_PTS * 4);
  float* v    = (float*)alloc(N_PTS * 4);
  float* objPart = (float*)alloc(256 * 4);
  unsigned int* scal = (unsigned int*)alloc(16 * 4);

  const dim3 b256(256);
  const dim3 gg(N_PTS / NTL, N_PTS / MTL);      // 16 x 16 GEMM blocks
  const dim3 hg(N_PTS / 16, N_PTS / 16);
  const int  nElemBlocks = (int)(NN / 256);

  k_init<<<1, 1, 0, stream>>>(scal);
  k_cvt_src<<<(N_PTS * D_EMB + 255) / 256, b256, 0, stream>>>(X, Xb);
  k_rownorm<<<N_PTS, b256, 0, stream>>>(X, sq);

  // C_src: Gram = X @ X^T, so Bt = X itself (N x K layout already)
  gemm_bf16<MODE_GRAM><<<gg, b256, 0, stream>>>(Xb, Xb, D_EMB, Df,
      nullptr, nullptr, nullptr, nullptr, sq, nullptr, nullptr, scal + 0);
  k_normalize<<<nElemBlocks, b256, 0, stream>>>(Df, scal + 0, C1b);

  // C_tgt: hyperbolic cost (f32 VALU; only 0.5 GFLOP, precision-critical)
  k_hyp<<<hg, b256, 0, stream>>>(H, Df, scal + 1);
  k_normalize<<<nElemBlocks, b256, 0, stream>>>(Df, scal + 1, C2b);

  k_rowmeansq<<<N_PTS, b256, 0, stream>>>(C1b, av);
  k_rowmeansq<<<N_PTS, b256, 0, stream>>>(C2b, bvv);
  k_fillT<<<nElemBlocks, b256, 0, stream>>>(Tb);

  for (int it = 0; it < GW_ITERS; ++it) {
    // TC2^T = (T @ C2^T)^T ; Bt = C2 (rows of C2 are columns of C2^T)
    gemm_bf16<MODE_STORET><<<gg, b256, 0, stream>>>(Tb, C2b, N_PTS,
        nullptr, TC2t, nullptr, nullptr, nullptr, nullptr, nullptr, nullptr, nullptr);
    // K = exp(-(2*constC - 4*C1@TC2)/eps), stored bf16 as K and K^T
    gemm_bf16<MODE_SINK><<<gg, b256, 0, stream>>>(C1b, TC2t, N_PTS,
        nullptr, Kb, KTb, av, bvv, nullptr, nullptr, nullptr, nullptr);
    // 50 Sinkhorn iterations in one persistent launch
    k_sinkhorn<<<SINK_BLOCKS, b256, 0, stream>>>(Kb, KTb, u, v, scal);
    // T = diag(u) K diag(v)
    k_formT<<<nElemBlocks, b256, 0, stream>>>(Kb, u, v, Tb, Df);
  }

  // objective: sum((constC - 2*C1@(T@C2^T)) * T)
  gemm_bf16<MODE_STORET><<<gg, b256, 0, stream>>>(Tb, C2b, N_PTS,
      nullptr, TC2t, nullptr, nullptr, nullptr, nullptr, nullptr, nullptr, nullptr);
  gemm_bf16<MODE_OBJ><<<gg, b256, 0, stream>>>(C1b, TC2t, N_PTS,
      nullptr, nullptr, nullptr, av, bvv, nullptr, Df, objPart, nullptr);
  k_finalsum<<<1, 1, 0, stream>>>(objPart, out);
}
